// AttentionAggregator_35210141892612
// MI455X (gfx1250) — compile-verified, matrix operand order checked by reading, not run on hardware
//
#include <hip/hip_runtime.h>
#include <hip/hip_bf16.h>
#include <stdint.h>

typedef __attribute__((ext_vector_type(16))) __bf16 v16bf;
typedef __attribute__((ext_vector_type(8)))  float  v8f;

#define DDIM 512
#define KNBR 32
#define TN   16

// ---- helpers -------------------------------------------------------------

__device__ __forceinline__ uint32_t f2bf(float x) {
  // round-to-nearest-even f32 -> bf16 (low 16 bits of result)
  uint32_t u = __builtin_bit_cast(uint32_t, x);
  return (u + 0x7FFFu + ((u >> 16) & 1u)) >> 16;
}
__device__ __forceinline__ uint32_t pack_bf2(float lo, float hi) {
  return f2bf(lo) | (f2bf(hi) << 16);
}

union Frag { uint4 q[2]; v16bf v; };

// VGPR/lane slot of element (m, k) of a 16-bit 16x32 WMMA A-matrix fragment
// (per CDNA5 ISA 7.12.2). B fragments use the mirrored layout (col <-> row),
// which the weight prepack below produces directly.
__device__ __forceinline__ void a_slot(int k, int m, int& lane, int& v) {
  int half = (k >> 3) & 1;
  lane = m + (half << 4);
  v = ((k >> 4) << 2) + ((k & 7) >> 1);
}

// ---- weight prepack: f32 row-major -> bf16 WMMA B-fragment tiles ---------
// w1p: 32 ntiles x 16 ktiles x (32 lanes x 8 dwords)   (W1^T, 512x512)
// wap: 32 ntiles x 32 ktiles x (32 lanes x 8 dwords)   (Wa^T, 1024x512)

__global__ __launch_bounds__(256)
void prepack_weights(const float* __restrict__ W1, const float* __restrict__ Wa,
                     uint32_t* __restrict__ w1p, uint32_t* __restrict__ wap) {
  const int NW1 = 32 * 16 * 256;
  const int NWA = 32 * 32 * 256;
  int idx = blockIdx.x * 256 + threadIdx.x;
  if (idx < NW1) {
    int v = idx & 7, lane = (idx >> 3) & 31, tile = idx >> 8;
    int kt = tile & 15, nt = tile >> 4;
    int n = lane & 15, khalf = lane >> 4;
    int k = ((v >> 2) << 4) + ((v & 3) << 1) + (khalf << 3);
    const float* p = W1 + (size_t)(nt * 16 + n) * DDIM + kt * 32 + k;
    w1p[idx] = pack_bf2(p[0], p[1]);
  } else if (idx < NW1 + NWA) {
    int i2 = idx - NW1;
    int v = i2 & 7, lane = (i2 >> 3) & 31, tile = i2 >> 8;
    int kt = tile & 31, nt = tile >> 5;
    int n = lane & 15, khalf = lane >> 4;
    int k = ((v >> 2) << 4) + ((v & 3) << 1) + (khalf << 3);
    const float* p = Wa + (size_t)(nt * 16 + n) * (2 * DDIM) + kt * 32 + k;
    wap[i2] = pack_bf2(p[0], p[1]);
  }
}

// ---- fused attention-aggregate kernel ------------------------------------

__global__ __launch_bounds__(256)
void attn_agg_kernel(const float* __restrict__ prev,
                     const float* __restrict__ neigh,
                     const unsigned char* __restrict__ mask,
                     const uint32_t* __restrict__ w1p,
                     const uint32_t* __restrict__ wap,
                     float* __restrict__ out) {
  // concat operand in WMMA A-fragment layout: kt 0..15 = agg, kt 16..31 = prev
  __shared__ __align__(16) uint32_t sCa[32][32][8];   // 32 KB
  __shared__ float sXf[TN][DDIM];                     // 32 KB (x = prev @ W1^T, f32)
  __shared__ float sS[TN][KNBR];
  __shared__ float sAttn[TN][KNBR];

  const int tid  = threadIdx.x;
  const int wave = tid >> 5, lane = tid & 31;
  const int nBase = blockIdx.x * TN;

  // ---- load prev tile once, convert to bf16 fragments (sCa[16..31]) ----
  #pragma unroll
  for (int i = 0; i < 16; ++i) {
    int p = tid + (i << 8);                  // 0..4095 pair-slots
    int m = p >> 8, dp = p & 255, d = dp << 1;
    float2 val = *(const float2*)(prev + (size_t)(nBase + m) * DDIM + d);
    int k = d & 31, kt = d >> 5, ls, vs;
    a_slot(k, m, ls, vs);
    sCa[16 + kt][ls][vs] = pack_bf2(val.x, val.y);
  }
  __syncthreads();

  // ---- stage A: X = P @ W1^T  (bf16 WMMA, f32 accumulate) ----
  for (int nti = 0; nti < 4; ++nti) {
    int nt = (wave << 2) + nti;
    v8f acc = {};
    #pragma unroll 4
    for (int kt = 0; kt < 16; ++kt) {
      Frag a, b;
      const uint4* ap = (const uint4*)&sCa[16 + kt][lane][0];
      a.q[0] = ap[0]; a.q[1] = ap[1];
      const uint4* bp = (const uint4*)(w1p + ((((nt << 4) + kt) << 8) + (lane << 3)));
      b.q[0] = bp[0]; b.q[1] = bp[1];
      acc = __builtin_amdgcn_wmma_f32_16x16x32_bf16(
          false, a.v, false, b.v, (short)0, acc, false, false);
    }
    int mb = (lane < 16) ? 0 : 8, nl = lane & 15;
    #pragma unroll
    for (int r = 0; r < 8; ++r) sXf[mb + r][(nt << 4) + nl] = acc[r];
  }
  __syncthreads();

  // ---- stage B: scores[n][k] = neigh[n,k,:] . x[n,:]  (f32, coalesced) ----
  for (int j = 0; j < 64; ++j) {
    int pr = (wave << 6) + j;
    int n = pr >> 5, k = pr & 31;
    const float4* np = (const float4*)(neigh + ((size_t)(nBase + n) * KNBR + k) * DDIM);
    const float4* xp = (const float4*)&sXf[n][0];
    if (j + 1 < 64) {   // warm L2/WGP$ for next neighbor row
      int p2 = pr + 1;
      __builtin_prefetch(neigh + ((size_t)(nBase + (p2 >> 5)) * KNBR + (p2 & 31)) * DDIM
                             + (lane << 4), 0, 3);
    }
    float4 acc = {0.f, 0.f, 0.f, 0.f};
    #pragma unroll
    for (int i = 0; i < 4; ++i) {
      int c = (i << 5) + lane;
      float4 aq = np[c], xq = xp[c];
      acc.x += aq.x * xq.x; acc.y += aq.y * xq.y;
      acc.z += aq.z * xq.z; acc.w += aq.w * xq.w;
    }
    float s = (acc.x + acc.y) + (acc.z + acc.w);
    #pragma unroll
    for (int off = 16; off >= 1; off >>= 1) s += __shfl_xor(s, off, 32);
    if (lane == 0) sS[n][k] = s;
  }
  __syncthreads();

  // ---- stage C: masked softmax over K=32 (one wave row-pair) ----
  #pragma unroll
  for (int r = 0; r < 2; ++r) {
    int n = (wave << 1) + r;
    float s = sS[n][lane];
    if (mask[(size_t)(nBase + n) * KNBR + lane]) s = -__builtin_inff();
    float mx = s;
    #pragma unroll
    for (int off = 16; off >= 1; off >>= 1) mx = fmaxf(mx, __shfl_xor(mx, off, 32));
    float e = __expf(s - mx);
    float sum = e;
    #pragma unroll
    for (int off = 16; off >= 1; off >>= 1) sum += __shfl_xor(sum, off, 32);
    sAttn[n][lane] = e / sum;
  }
  __syncthreads();

  // ---- stage D: agg[n,:] = attn[n,:] @ neigh[n,:,:]  -> sCa[0..15] bf16 ----
  #pragma unroll
  for (int r = 0; r < 2; ++r) {
    int m = (wave << 1) + r;
    const float* nrow = neigh + (size_t)(nBase + m) * KNBR * DDIM;
    #pragma unroll
    for (int i = 0; i < 4; ++i) {
      int d4 = (((i << 5) + lane) << 2);
      float4 acc = {0.f, 0.f, 0.f, 0.f};
      for (int k = 0; k < KNBR; ++k) {
        float a = sAttn[m][k];
        float4 vv = *(const float4*)(nrow + k * DDIM + d4);
        acc.x += a * vv.x; acc.y += a * vv.y;
        acc.z += a * vv.z; acc.w += a * vv.w;
      }
      int k0 = d4 & 31, kt = d4 >> 5, ls, vs;
      a_slot(k0, m, ls, vs);
      sCa[kt][ls][vs] = pack_bf2(acc.x, acc.y);
      a_slot(k0 + 2, m, ls, vs);
      sCa[kt][ls][vs] = pack_bf2(acc.z, acc.w);
    }
  }
  __syncthreads();

  // ---- stage E: out = tanh(concat([agg, prev]) @ Wa^T)  (bf16 WMMA) ----
  for (int nti = 0; nti < 4; ++nti) {
    int nt = (wave << 2) + nti;
    v8f acc = {};
    #pragma unroll 4
    for (int kt = 0; kt < 32; ++kt) {
      Frag a, b;
      const uint4* ap = (const uint4*)&sCa[kt][lane][0];
      a.q[0] = ap[0]; a.q[1] = ap[1];
      const uint4* bp = (const uint4*)(wap + ((((nt << 5) + kt) << 8) + (lane << 3)));
      b.q[0] = bp[0]; b.q[1] = bp[1];
      acc = __builtin_amdgcn_wmma_f32_16x16x32_bf16(
          false, a.v, false, b.v, (short)0, acc, false, false);
    }
    int mb = (lane < 16) ? 0 : 8, nl = lane & 15;
    #pragma unroll
    for (int r = 0; r < 8; ++r)
      out[(size_t)(nBase + mb + r) * DDIM + (nt << 4) + nl] = tanhf(acc[r]);
  }
}

// ---- launcher ------------------------------------------------------------

extern "C" void kernel_launch(void* const* d_in, const int* in_sizes, int n_in,
                              void* d_out, int out_size, void* d_ws, size_t ws_size,
                              hipStream_t stream) {
  const float*         prev  = (const float*)d_in[0];
  const float*         neigh = (const float*)d_in[1];
  const unsigned char* mask  = (const unsigned char*)d_in[2];
  const float*         W1    = (const float*)d_in[3];
  const float*         Wa    = (const float*)d_in[4];
  float*               out   = (float*)d_out;

  uint32_t* w1p = (uint32_t*)d_ws;                 // 512 KB
  uint32_t* wap = w1p + 32 * 16 * 256;             // 1 MB

  const int N = in_sizes[0] / DDIM;                // 20000
  const int total = 32 * 16 * 256 + 32 * 32 * 256; // prepack dwords

  prepack_weights<<<(total + 255) / 256, 256, 0, stream>>>(W1, Wa, w1p, wap);
  attn_agg_kernel<<<N / TN, 256, 0, stream>>>(prev, neigh, mask, w1p, wap, out);
}
